// SimpleSpConvNet_90022514524741
// MI455X (gfx1250) — matrix operand order repaired and checked by reading, not compiled, not run on hardware
//
#include <hip/hip_runtime.h>

// ---------------------------------------------------------------------------
// CDNA5 (gfx1250) sparse-conv GGS pipeline using v_wmma_f32_16x16x32_f16.
// wave32: one wave handles 4 consecutive 16-pair tiles for one offset k,
// keeping the WMMA B-operand (weights) resident in registers across tiles.
// ---------------------------------------------------------------------------

typedef __attribute__((ext_vector_type(16))) _Float16 v16h;
typedef __attribute__((ext_vector_type(8)))  float    v8f;

union Frag {
    v16h h;
    uint4 q[2];
};

// ---------------------------------------------------------------------------
// Prep: fp32 feats -> f16 rows (16 ch), extra zero row at index N
// ---------------------------------------------------------------------------
__global__ void feats_to_half(const float* __restrict__ f, _Float16* __restrict__ fh,
                              int nrows /* N+1 */) {
    unsigned i = blockIdx.x * blockDim.x + threadIdx.x;
    unsigned total = (unsigned)nrows * 16u;
    if (i >= total) return;
    unsigned row = i >> 4;
    fh[i] = (row + 1 < (unsigned)nrows) ? (_Float16)f[i] : (_Float16)0.0f;
}

// ---------------------------------------------------------------------------
// Prep: pack W[k][Cin][Cout] (f32) into WMMA B-operand chunks (f16).
// Chunk = (k*KB + kb)*NT + nt ; per chunk: 32 lanes x 16 halves.
// B (K x N = 32 x 16): lanes 0-15 hold K=0..7,16..23 (col = lane&15),
// lanes 16-31 hold K=8..15,24..31; half j<8 -> K=8g+j, j>=8 -> K=16+8g+(j-8).
// ---------------------------------------------------------------------------
__global__ void pack_w(const float* __restrict__ W, int Cin, int Cout,
                       int KBk, int NTk, _Float16* __restrict__ dst) {
    int i = blockIdx.x * blockDim.x + threadIdx.x;
    int total = 27 * KBk * NTk * 512;
    if (i >= total) return;
    int j     = i & 15;
    int lane  = (i >> 4) & 31;
    int chunk = i >> 9;
    int nt = chunk % NTk;
    int kb = (chunk / NTk) % KBk;
    int k  = chunk / (NTk * KBk);
    int g = lane >> 4, n = lane & 15;
    int cin  = kb * 32 + 8 * g + (j & 7) + ((j >> 3) << 4);
    int cout = nt * 16 + n;
    float v = 0.0f;
    if (cin < Cin && cout < Cout)
        v = W[((size_t)k * Cin + cin) * Cout + cout];
    dst[i] = (_Float16)v;
}

__global__ void zero_f32(float* __restrict__ p, size_t n) {
    size_t i = (size_t)blockIdx.x * blockDim.x + threadIdx.x;
    if (i < n) p[i] = 0.0f;
}

// ---------------------------------------------------------------------------
// Gather -> WMMA -> scatter-atomic GGS kernel.
//   KB   : K32 blocks       (1: Cin=16 zero-padded, 2: Cin=64)
//   NT   : 16-wide N tiles  (4: Cout=64, 2: Cout=32)
//   OST  : output row stride (elements)  -- compile-time for cheap addressing
//   SSTR : source row stride (halves)    -- compile-time for cheap addressing
// One wave processes TILES consecutive 16-row tiles for one k (blockIdx.y);
// weight fragments are loaded once per wave and reused across tiles.
// ---------------------------------------------------------------------------
template <int KB, int NT, int OST, int SSTR>
__global__ void ggs_kernel(const _Float16* __restrict__ src,
                           const int* __restrict__ pin, const int* __restrict__ pout,
                           int P,
                           const _Float16* __restrict__ Wp,
                           const int* __restrict__ n_out_dev, int n_out_const,
                           int zero_row,
                           float* __restrict__ out) {
    constexpr int TILES = 4;
    const int lane = threadIdx.x & 31;
    const int wave = threadIdx.x >> 5;
    const int tile0 = (blockIdx.x * (blockDim.x >> 5) + wave) * TILES;
    const int k     = blockIdx.y;
    if (tile0 * 16 >= P) return;                 // wave-uniform

    const int n_out = n_out_dev ? n_out_dev[0] : n_out_const;
    const int g = lane >> 4;                     // half-wave group
    const int m = lane & 15;                     // column / row-in-tile

    const int* pin_k  = pin  + (size_t)k * P;
    const int* pout_k = pout + (size_t)k * P;

    // ---- hoist B (weight) fragments: reused across all tiles of this wave ----
    Frag B[KB * NT];
    {
        const _Float16* wbase = Wp + (size_t)k * (KB * NT) * 512 + lane * 16;
#pragma unroll
        for (int q = 0; q < KB * NT; ++q) {
            const _Float16* bp = wbase + q * 512;
            B[q].q[0] = *(const uint4*)bp;
            B[q].q[1] = *(const uint4*)(bp + 8);
        }
    }

#pragma unroll
    for (int t = 0; t < TILES; ++t) {
        const int t16 = (tile0 + t) * 16;
        if (t16 >= P) break;                     // wave-uniform

        // gather index for this lane's A row (pad/out-of-range -> zeroed row)
        const int grow = t16 + m;
        const int gi = (grow < P) ? pin_k[grow] : zero_row;
        const _Float16* srow = src + (unsigned)gi * (unsigned)SSTR;

        // scatter indices for this lane's 8 output rows
        int po[8];
#pragma unroll
        for (int r = 0; r < 8; ++r) {
            int orow = t16 + 8 * g + r;
            po[r] = (orow < P) ? pout_k[orow] : 0x7fffffff;
        }

        // A fragments: lanes 0-15 take ch [c0..c0+7] + [c0+16..c0+23],
        // lanes 16-31 take ch [c0+8..c0+15] + [c0+24..c0+31] (ISA A layout)
        Frag A[KB];
#pragma unroll
        for (int kb = 0; kb < KB; ++kb) {
            const _Float16* p0 = srow + kb * 32 + g * 8;
            A[kb].q[0] = *(const uint4*)p0;
            if (KB == 1) {
                A[kb].q[1] = make_uint4(0u, 0u, 0u, 0u);  // zero-padded K 16..31
            } else {
                A[kb].q[1] = *(const uint4*)(p0 + 16);
            }
        }

#pragma unroll
        for (int nt = 0; nt < NT; ++nt) {
            v8f c = {};
#pragma unroll
            for (int kb = 0; kb < KB; ++kb) {
                c = __builtin_amdgcn_wmma_f32_16x16x32_f16(
                        false, A[kb].h, false, B[kb * NT + nt].h, (short)0, c,
                        false, false);
            }
            // scatter-add: lane holds column m, rows 8*g + r (32-bit offsets)
#pragma unroll
            for (int r = 0; r < 8; ++r) {
                if (po[r] < n_out) {
                    unsigned off = (unsigned)po[r] * (unsigned)OST
                                 + (unsigned)(nt * 16 + m);
                    atomicAdd(out + off, c[r]);
                }
            }
        }
    }
}

// ---------------------------------------------------------------------------
// BatchNorm: per-channel sum / sumsq over rows [0, n1)
// ---------------------------------------------------------------------------
__global__ void bn_stats(const float* __restrict__ x, const int* __restrict__ n1p,
                         float* __restrict__ sums /* [0..63]=sum, [64..127]=sumsq */) {
    const int n1 = n1p[0];
    const int c  = threadIdx.x & 63;
    const int rg = threadIdx.x >> 6;            // 0..3
    float s = 0.0f, s2 = 0.0f;
    for (int row = blockIdx.x * 4 + rg; row < n1; row += gridDim.x * 4) {
        float v = x[(unsigned)row * 64u + c];
        s += v;
        s2 += v * v;
    }
    atomicAdd(&sums[c], s);
    atomicAdd(&sums[64 + c], s2);
}

// BN(train-stats) + ReLU -> f16; rows >= n1 (up to rows_total-1) written as 0
__global__ void bn_apply(const float* __restrict__ x, const float* __restrict__ sums,
                         const float* __restrict__ gamma, const float* __restrict__ beta,
                         const int* __restrict__ n1p, int rows_total /* P2+1 */,
                         _Float16* __restrict__ y) {
    const int n1 = n1p[0];
    const int c  = threadIdx.x & 63;
    const int rg = threadIdx.x >> 6;
    const int row = blockIdx.x * 4 + rg;
    if (row >= rows_total) return;
    float outv = 0.0f;
    if (row < n1) {
        const float inv_n = 1.0f / (float)n1;
        const float mean  = sums[c] * inv_n;
        const float var   = sums[64 + c] * inv_n - mean * mean;
        const float sc    = gamma[c] * rsqrtf(var + 1e-5f);
        const float bi    = beta[c] - mean * sc;
        float v = x[(unsigned)row * 64u + c] * sc + bi;
        outv = v > 0.0f ? v : 0.0f;
    }
    y[(unsigned)row * 64u + c] = (_Float16)outv;
}

// ---------------------------------------------------------------------------
// Host launch
// ---------------------------------------------------------------------------
extern "C" void kernel_launch(void* const* d_in, const int* in_sizes, int n_in,
                              void* d_out, int out_size, void* d_ws, size_t ws_size,
                              hipStream_t stream) {
    const float* feats  = (const float*)d_in[0];
    const float* W1     = (const float*)d_in[1];
    const float* gamma1 = (const float*)d_in[2];
    const float* beta1  = (const float*)d_in[3];
    const float* W2     = (const float*)d_in[4];
    const float* gamma2 = (const float*)d_in[5];
    const float* beta2  = (const float*)d_in[6];
    const float* W3     = (const float*)d_in[7];
    const int* p1_in  = (const int*)d_in[8];
    const int* p1_out = (const int*)d_in[9];
    const int* p2_in  = (const int*)d_in[10];
    const int* p2_out = (const int*)d_in[11];
    const int* n1p    = (const int*)d_in[12];

    const int N  = in_sizes[0] / 16;       // 200000
    const int P1 = in_sizes[8] / 27;
    const int P2 = in_sizes[10] / 27;      // >= n1 (center offset pairs every site)

    // ---- carve workspace ----
    char* w = (char*)d_ws;
    auto alloc = [&](size_t bytes) -> void* {
        void* p = (void*)w;
        w += (bytes + 255) & ~(size_t)255;
        return p;
    };
    _Float16* featsH = (_Float16*)alloc((size_t)(N + 1) * 16 * sizeof(_Float16));
    _Float16* W1p    = (_Float16*)alloc((size_t)27 * 1 * 4 * 512 * sizeof(_Float16));
    _Float16* W2p    = (_Float16*)alloc((size_t)27 * 2 * 4 * 512 * sizeof(_Float16));
    _Float16* W3p    = (_Float16*)alloc((size_t)27 * 2 * 2 * 512 * sizeof(_Float16));
    float*    acc    = (float*)   alloc((size_t)(P2 + 1) * 64 * sizeof(float));
    _Float16* xh     = (_Float16*)alloc((size_t)(P2 + 1) * 64 * sizeof(_Float16));
    float*    stats  = (float*)   alloc(128 * sizeof(float));

    const int BT = 256;                    // 8 waves per block
    const int TILES = 4;                   // tiles per wave (must match kernel)
    auto cdiv = [](size_t a, size_t b) { return (unsigned)((a + b - 1) / b); };

    // ---- prep: feats f16 + packed weights ----
    feats_to_half<<<cdiv((size_t)(N + 1) * 16, BT), BT, 0, stream>>>(feats, featsH, N + 1);
    pack_w<<<cdiv(27 * 1 * 4 * 512, BT), BT, 0, stream>>>(W1, 16, 64, 1, 4, W1p);
    pack_w<<<cdiv(27 * 2 * 4 * 512, BT), BT, 0, stream>>>(W2, 64, 64, 2, 4, W2p);
    pack_w<<<cdiv(27 * 2 * 2 * 512, BT), BT, 0, stream>>>(W3, 64, 32, 2, 2, W3p);

    const size_t accN = (size_t)(P2 + 1) * 64;
    zero_f32<<<cdiv(accN, BT), BT, 0, stream>>>(acc, accN);
    zero_f32<<<1, 128, 0, stream>>>(stats, 128);

    // ---- stage 1: SparseConv3d 16 -> 64 ----
    {
        unsigned tiles = (unsigned)((P1 + 15) / 16);
        dim3 grid((tiles + 8 * TILES - 1) / (8 * TILES), 27);
        ggs_kernel<1, 4, 64, 16><<<grid, BT, 0, stream>>>(featsH, p1_in, p1_out, P1,
                                                          W1p, n1p, 0, N, acc);
    }
    bn_stats<<<4096, 256, 0, stream>>>(acc, n1p, stats);
    bn_apply<<<cdiv((size_t)P2 + 1, 4), 256, 0, stream>>>(acc, stats, gamma1, beta1,
                                                          n1p, P2 + 1, xh);

    zero_f32<<<cdiv(accN, BT), BT, 0, stream>>>(acc, accN);
    zero_f32<<<1, 128, 0, stream>>>(stats, 128);

    // ---- stage 2: SubMConv3d 64 -> 64 ----
    {
        unsigned tiles = (unsigned)((P2 + 15) / 16);
        dim3 grid((tiles + 8 * TILES - 1) / (8 * TILES), 27);
        ggs_kernel<2, 4, 64, 64><<<grid, BT, 0, stream>>>(xh, p2_in, p2_out, P2,
                                                          W2p, n1p, 0, P2, acc);
    }
    bn_stats<<<4096, 256, 0, stream>>>(acc, n1p, stats);
    bn_apply<<<cdiv((size_t)P2 + 1, 4), 256, 0, stream>>>(acc, stats, gamma2, beta2,
                                                          n1p, P2 + 1, xh);

    // ---- stage 3: SparseInverseConv3d 64 -> 32 (pairs swapped) ----
    float* outp = (float*)d_out;
    zero_f32<<<cdiv((size_t)N * 32, BT), BT, 0, stream>>>(outp, (size_t)N * 32);
    {
        unsigned tiles = (unsigned)((P1 + 15) / 16);
        dim3 grid((tiles + 8 * TILES - 1) / (8 * TILES), 27);
        ggs_kernel<2, 2, 32, 64><<<grid, BT, 0, stream>>>(xh, p1_out, p1_in, P1,
                                                          W3p, nullptr, N, P2, outp);
    }
}
